// GraphAttentionLayer_19439021981809
// MI455X (gfx1250) — matrix operand order repaired and checked by reading, not compiled, not run on hardware
//
#include <hip/hip_runtime.h>

// GAT layer for MI455X (gfx1250), wave32 + WMMA + TDM/async-LDS staging.
// Kernel 1: hp = h @ Wcat via v_wmma_f32_16x16x32_f16, fused e_src/e_dst epilogue.
// Kernel 2: fused dense-softmax attention (single pass: row max is closed-form
//           because leakyrelu is monotonic) + att@hp WMMA + residual + ELU.
//           V chunks staged into LDS via TENSOR_LOAD_TO_LDS (TDM) when the
//           builtin exists, else GLOBAL_LOAD_ASYNC_TO_LDS_B128, else manual.
// Workspace: hp16 (f16, 4MB) | e_src (f32 128KB) | e_dst (f32 128KB).

#define LRELU_ALPHA 0.2f

typedef __attribute__((ext_vector_type(16))) _Float16 v16h;
typedef __attribute__((ext_vector_type(8)))  float    v8f;
typedef __attribute__((ext_vector_type(4)))  unsigned int v4u;
typedef __attribute__((ext_vector_type(8)))  int      v8i;
typedef __attribute__((ext_vector_type(4)))  int      v4i;

#if __has_builtin(__builtin_amdgcn_tensor_load_to_lds) && __has_builtin(__builtin_amdgcn_s_wait_tensorcnt)
#define USE_TDM 1
#else
#define USE_TDM 0
#endif
#if __has_builtin(__builtin_amdgcn_global_load_async_to_lds_b32) && \
    __has_builtin(__builtin_amdgcn_global_load_async_to_lds_b128) && \
    __has_builtin(__builtin_amdgcn_s_wait_asynccnt)
#define USE_ASYNC 1
#else
#define USE_ASYNC 0
#endif

// Toolchain's async-LDS builtins take plain generic int* params
// (per hipcc diagnostic); the backend resolves global vs LDS sides.
#define TO_GLOBAL(p) ((int*)(p))
#define TO_LDS(p)    ((int*)(p))

constexpr int B_   = 16;
constexpr int N_   = 1024;
constexpr int IND  = 128;   // in_dim == out_dim (heads*head_dim)
constexpr int H_   = 2;
constexpr int HD   = 64;
constexpr int TILE = 128;

// padded LDS leading dims (elements) to dodge bank conflicts on the
// 16-lane row-strided WMMA fragment loads
constexpr int LDH = 136;  // halfs, kernel 1 (sH / sWt)
constexpr int LDC = 132;  // floats, kernel 1 (sHP)
constexpr int LDP = 136;  // halfs, kernel 2 (sP / sV)
constexpr int LDO = 68;   // floats, kernel 2 (sO)

__device__ __forceinline__ float lrelu(float x) { return x >= 0.f ? x : LRELU_ALPHA * x; }

// A-matrix 16x32 f16 fragment (ISA 7.12.2): lane&15 = M row, lane>>4 = group g.
// VGPR j holds K pair: k = (j>>2)*16 + g*8 + (j&3)*2  (pairs contiguous in memory).
__device__ __forceinline__ v16h load_a_frag(const _Float16* __restrict__ base, int ld,
                                            int row, int g, int kk) {
  const _Float16* rp = base + row * ld + kk;
  v16h a;
  unsigned int* au = reinterpret_cast<unsigned int*>(&a);
#pragma unroll
  for (int j = 0; j < 8; ++j) {
    int k = ((j >> 2) << 4) + (g << 3) + ((j & 3) << 1);
    au[j] = *reinterpret_cast<const unsigned int*>(rp + k);
  }
  return a;
}

// B-matrix 32x16 f16 fragment: lane&15 = N col, lane>>4 = group g;
// VGPR j holds K pair k = (j>>2)*16 + g*8 + (j&3)*2 (same as A). Requires K-contiguous
// (d-major) storage in LDS.
__device__ __forceinline__ v16h load_b_frag(const _Float16* __restrict__ base, int ld,
                                            int col, int g, int kk) {
  const _Float16* cp = base + col * ld + kk + (g << 3);
  v16h b;
  unsigned int* bu = reinterpret_cast<unsigned int*>(&b);
#pragma unroll
  for (int j = 0; j < 8; ++j)
    bu[j] = *reinterpret_cast<const unsigned int*>(cp + ((j >> 2) << 4) + ((j & 3) << 1));
  return b;
}

// ---------------------------------------------------------------------------
// Kernel 1: hp tile GEMM (128 rows x 128 cols, K=128) + e_src/e_dst epilogue
// ---------------------------------------------------------------------------
__global__ __launch_bounds__(256)
void gat_hp_wmma(const float* __restrict__ hg, const float* __restrict__ Wg,
                 const float* __restrict__ ag,
                 _Float16* __restrict__ hp16,
                 float* __restrict__ e_src, float* __restrict__ e_dst) {
  extern __shared__ __align__(16) char smem[];
  _Float16* sH  = reinterpret_cast<_Float16*>(smem);          // 128 x LDH
  _Float16* sWt = sH + TILE * LDH;                            // 128 x LDH ([c][i])
  float*    sHP = reinterpret_cast<float*>(sWt + TILE * LDH); // 128 x LDC

  const int tid  = threadIdx.x;
  const int row0 = blockIdx.x * TILE;

  // Stage Wcat transposed: column c -> head c>>6, d = c&63.  W is (H, IND, HD).
#pragma unroll 4
  for (int idx = tid; idx < TILE * IND; idx += 256) {
    int c = idx >> 7, i = idx & 127;
    float w = Wg[(c >> 6) * (IND * HD) + i * HD + (c & 63)];
    sWt[c * LDH + i] = (_Float16)w;
  }
  // Stage h tile, f32 -> f16, vectorized float4 loads.
#pragma unroll
  for (int j = 0; j < 16; ++j) {
    int linear = (tid << 2) + j * 1024;
    int r = linear >> 7, i = linear & 127;
    const float4 v = *reinterpret_cast<const float4*>(&hg[(size_t)(row0 + r) * IND + i]);
    sH[r * LDH + i + 0] = (_Float16)v.x;
    sH[r * LDH + i + 1] = (_Float16)v.y;
    sH[r * LDH + i + 2] = (_Float16)v.z;
    sH[r * LDH + i + 3] = (_Float16)v.w;
  }
  __syncthreads();

  const int lane = tid & 31;
  const int wv   = tid >> 5;        // 8 waves: wave w owns row strip w*16
  const int g    = lane >> 4;
  const int mrow = wv * 16 + (lane & 15);

  v16h afr[4];
#pragma unroll
  for (int kc = 0; kc < 4; ++kc) afr[kc] = load_a_frag(sH, LDH, mrow, g, kc * 32);

#pragma unroll
  for (int ct = 0; ct < 8; ++ct) {
    // gather all K-step B fragments first, then run the WMMA chain back-to-back
    v16h bfr[4];
#pragma unroll
    for (int kc = 0; kc < 4; ++kc)
      bfr[kc] = load_b_frag(sWt, LDH, ct * 16 + (lane & 15), g, kc * 32);
    v8f acc = {};
#pragma unroll
    for (int kc = 0; kc < 4; ++kc)
      acc = __builtin_amdgcn_wmma_f32_16x16x32_f16(false, afr[kc], false, bfr[kc],
                                                   (short)0, acc, false, false);
    int col = ct * 16 + (lane & 15);
#pragma unroll
    for (int r = 0; r < 8; ++r)                     // C/D layout: M = r + 8*g
      sHP[(wv * 16 + r + (g << 3)) * LDC + col] = acc[r];
  }
  __syncthreads();

  // Epilogue A: e_src/e_dst = hp(head slice) . a_src/a_dst. 2 threads/row.
  {
    int head = tid >> 7;
    int r    = tid & 127;
    int gn   = row0 + r;
    int b    = gn >> 10;
    int n    = gn & 1023;
    const float* arow = ag + head * (2 * HD);
    float es = 0.f, ed = 0.f;
#pragma unroll 8
    for (int d = 0; d < HD; ++d) {
      float v = sHP[r * LDC + head * HD + d];
      es += v * arow[d];
      ed += v * arow[HD + d];
    }
    e_src[(b * H_ + head) * N_ + n] = es;
    e_dst[(b * H_ + head) * N_ + n] = ed;
  }

  // Epilogue B: hp -> f16 global (packed 2-half stores).
  unsigned int* hpu = reinterpret_cast<unsigned int*>(hp16);
#pragma unroll 4
  for (int j = 0; j < 32; ++j) {
    int linear = (tid << 1) + j * 512;
    int r = linear >> 7, c = linear & 127;
    union { _Float16 hh[2]; unsigned int u; } pk;
    pk.hh[0] = (_Float16)sHP[r * LDC + c];
    pk.hh[1] = (_Float16)sHP[r * LDC + c + 1];
    hpu[((unsigned)(row0 + r) * IND + c) >> 1] = pk.u;
  }
}

// ---------------------------------------------------------------------------
// Kernel 2: fused softmax(leakyrelu(e_src[n]+e_dst[m])) @ hp  + residual + ELU
// One block per (b, head, 128-row n-tile).
// ---------------------------------------------------------------------------
__global__ __launch_bounds__(256)
void gat_attn_wmma(const float* __restrict__ hg, const _Float16* __restrict__ hp16,
                   const float* __restrict__ e_src, const float* __restrict__ e_dst,
                   float* __restrict__ out) {
  extern __shared__ __align__(16) char smem[];
  _Float16* sP    = reinterpret_cast<_Float16*>(smem); // 128 x LDP  (P tile, f16)
  _Float16* sV    = sP + TILE * LDP;                   // 64 x LDP   (V, d-major)
  _Float16* sVraw = sV + HD * LDP;                     // 128 x 64   (V, m-major stage)
  float*    sO    = reinterpret_cast<float*>(sVraw + TILE * HD); // 128 x LDO
  float*    sEs   = sO + TILE * LDO;   // 128
  float*    sRM   = sEs + TILE;        // 128 row max
  float*    sRS   = sRM + TILE;        // 128 row sum
  float*    sEd   = sRS + TILE;        // 128 e_dst chunk
  float*    sRed  = sEd + TILE;        // 256 reduction scratch

  const int tid = threadIdx.x;
  const int bh  = blockIdx.x >> 3;   // b*2 + head
  const int nt  = blockIdx.x & 7;
  const int b   = bh >> 1;
  const int hd  = bh & 1;

  // Closed-form softmax max: global max of e_dst over all m for this (b,head).
  float v = -3.402823466e38f;
  for (int i = tid; i < N_; i += 256) v = fmaxf(v, e_dst[bh * N_ + i]);
  sRed[tid] = v;
  __syncthreads();
#pragma unroll
  for (int s = 128; s > 0; s >>= 1) {
    if (tid < s) sRed[tid] = fmaxf(sRed[tid], sRed[tid + s]);
    __syncthreads();
  }
  const float dmax = sRed[0];

  if (tid < TILE) {
    float es = e_src[bh * N_ + nt * TILE + tid];
    sEs[tid] = es;
    sRM[tid] = lrelu(es + dmax);
    sRS[tid] = 0.f;
  }

  const int lane = tid & 31;
  const int wv   = tid >> 5;
  const int g    = lane >> 4;

  v8f acc[4] = {};   // persistent O accumulators (16 rows x 64 cols per wave)

  for (int mc = 0; mc < 8; ++mc) {
    const int m0 = mc * TILE;
    __syncthreads();  // previous iteration's WMMA reads of sP/sV/sEd done

    // ---- stage e_dst chunk (128 f32) -------------------------------------
#if USE_ASYNC
    if (tid < TILE)  // waves 0-3 fully active: one b32 async copy per lane
      __builtin_amdgcn_global_load_async_to_lds_b32(
          TO_GLOBAL(&e_dst[bh * N_ + m0 + tid]), TO_LDS(&sEd[tid]), 0, 0);
#else
    if (tid < TILE) sEd[tid] = e_dst[bh * N_ + m0 + tid];
#endif

    // ---- stage V chunk (128 m x 64 d f16) into sVraw, row-major ----------
    const _Float16* gvbase = &hp16[((size_t)(b * N_ + m0)) * IND + hd * HD];
#if USE_TDM
    if (wv == 0) {
      // Tensor DMA descriptor (ISA 8.3/8.4): 2D tile 64(d) x 128(m), 2B elems,
      // row stride 128 elems; destination = sVraw (dense 64-elem rows).
      unsigned long long ga = (unsigned long long)gvbase;
      v4u g0;
      g0.x = 1u;                                        // count=1, no gather
      g0.y = (unsigned int)(unsigned long long)sVraw;   // lds_addr (bytes)
      g0.z = (unsigned int)ga;                          // global_addr[31:0]
      g0.w = (unsigned int)((ga >> 32) & 0x1FFFFFFull)  // global_addr[56:32]
           | (2u << 30);                                // type=2 (image)
      v8i g1;
      g1[0] = (int)(1u << 16);          // workgroup_mask=0, data_size=1 (2B)
      g1[1] = (int)(128u << 16);        // tensor_dim0 = 128 (low16 in [31:16])
      g1[2] = (int)(128u << 16);        // tensor_dim0 hi=0 | tensor_dim1=128
      g1[3] = (int)(64u << 16);         // tensor_dim1 hi=0 | tile_dim0=64
      g1[4] = (int)128u;                // tile_dim1=128, tile_dim2=0
      g1[5] = (int)128u;                // tensor_dim0_stride = 128
      g1[6] = 0;                        // stride hi / dim1_stride lo
      g1[7] = 0;
      v4i gz = {0, 0, 0, 0};
#if __clang_major__ >= 23
      v8i gz8 = {0, 0, 0, 0, 0, 0, 0, 0};
      __builtin_amdgcn_tensor_load_to_lds(g0, g1, gz, gz, gz8, 0);
#else
      __builtin_amdgcn_tensor_load_to_lds(g0, g1, gz, gz, 0);
#endif
      __builtin_amdgcn_s_wait_tensorcnt(0);
    }
#elif USE_ASYNC
    // async DMA: 1024 lanes-iters x 16B, 8 x b128 per 128B row slice
#pragma unroll
    for (int j = 0; j < 4; ++j) {
      int linear = tid + j * 256;
      int mi  = linear >> 3;
      int off = (linear & 7) << 4;   // byte offset within the 128B row slice
      __builtin_amdgcn_global_load_async_to_lds_b128(
          TO_GLOBAL((const char*)&gvbase[(size_t)mi * IND] + off),
          TO_LDS((char*)&sVraw[mi * HD] + off), 0, 0);
    }
#else
#pragma unroll
    for (int j = 0; j < 4; ++j) {
      int linear = tid + j * 256;
      int mi  = linear >> 3;
      int off = (linear & 7) << 2;   // uint4 index within the 32-uint row slice
      *reinterpret_cast<uint4*>(reinterpret_cast<unsigned int*>(&sVraw[mi * HD]) + off) =
          *reinterpret_cast<const uint4*>(
              reinterpret_cast<const unsigned int*>(&gvbase[(size_t)mi * IND]) + off);
    }
#endif
    if (mc + 1 < 8)  // global_prefetch_b8 of next V chunk
      __builtin_prefetch(&hp16[((size_t)(b * N_ + m0 + TILE + (tid & 127))) * IND + hd * HD], 0, 1);
#if USE_ASYNC
    __builtin_amdgcn_s_wait_asynccnt(0);
#endif
    __syncthreads();

    // ---- transpose sVraw [m][d] -> sV [d][m] (K=m contiguous for B frags) -
#pragma unroll
    for (int j = 0; j < 16; ++j) {
      int linear = tid + j * 256;          // uint id in [0, 4096)
      int mi = linear >> 5;
      int d  = (linear & 31) << 1;
      unsigned int u = reinterpret_cast<const unsigned int*>(sVraw)[mi * 32 + (linear & 31)];
      union { unsigned int u; _Float16 hh[2]; } pk; pk.u = u;
      sV[(d    ) * LDP + mi] = pk.hh[0];
      sV[(d + 1) * LDP + mi] = pk.hh[1];
    }

    // ---- P = exp(lrelu(e_src+e_dst) - rowmax), f16; row sums --------------
    {
      int row = tid >> 1;
      int mb  = (tid & 1) << 6;
      float es = sEs[row];
      float rm = sRM[row];
      float psum = 0.f;
#pragma unroll 8
      for (int k = 0; k < 64; ++k) {
        float e = lrelu(es + sEd[mb + k]);
        float p = __expf(e - rm);
        psum += p;
        sP[row * LDP + mb + k] = (_Float16)p;
      }
      atomicAdd(&sRS[row], psum);   // ds_add_f32
    }
    __syncthreads();

    // ---- O += P(128x128) @ V(128x64), wave w owns rows [16w,16w+16) -------
    const int mrow = wv * 16 + (lane & 15);
    v16h afr[4];
#pragma unroll
    for (int kc = 0; kc < 4; ++kc) afr[kc] = load_a_frag(sP, LDP, mrow, g, kc * 32);
#pragma unroll
    for (int ct = 0; ct < 4; ++ct) {
      v16h bfr[4];
#pragma unroll
      for (int kc = 0; kc < 4; ++kc)
        bfr[kc] = load_b_frag(sV, LDP, ct * 16 + (lane & 15), g, kc * 32);
#pragma unroll
      for (int kc = 0; kc < 4; ++kc)
        acc[ct] = __builtin_amdgcn_wmma_f32_16x16x32_f16(false, afr[kc], false, bfr[kc],
                                                         (short)0, acc[ct], false, false);
    }
  }

  __syncthreads();
#pragma unroll
  for (int ct = 0; ct < 4; ++ct) {
    int d = ct * 16 + (lane & 15);
#pragma unroll
    for (int r = 0; r < 8; ++r)
      sO[(wv * 16 + r + (g << 3)) * LDO + d] = acc[ct][r];
  }
  __syncthreads();

  // Normalize by row sum, add residual h, ELU, store.
#pragma unroll 4
  for (int j = 0; j < 32; ++j) {
    int linear = tid + j * 256;
    int row = linear >> 6;
    int d   = linear & 63;
    int n   = nt * TILE + row;
    size_t gidx = ((size_t)(b * N_ + n)) * IND + hd * HD + d;
    float x = sO[row * LDO + d] / sRS[row] + hg[gidx];
    out[gidx] = x > 0.f ? x : expm1f(x);
  }
}

// ---------------------------------------------------------------------------
extern "C" void kernel_launch(void* const* d_in, const int* in_sizes, int n_in,
                              void* d_out, int out_size, void* d_ws, size_t ws_size,
                              hipStream_t stream) {
  const float* hg = (const float*)d_in[0];   // (16,1024,128)
  const float* Wg = (const float*)d_in[1];   // (2,128,64)
  const float* ag = (const float*)d_in[2];   // (2,128,1)
  float* out = (float*)d_out;                // (16,1024,128)

  _Float16* hp16 = (_Float16*)d_ws;                                  // 16384*128 halfs = 4MB
  float* e_src = (float*)((char*)d_ws + (size_t)16384 * 128 * 2);    // 32*1024 f32
  float* e_dst = e_src + 32 * 1024;

  constexpr size_t SMEM_A = (size_t)(TILE * LDH) * 2 * 2   // sH + sWt
                          + (size_t)(TILE * LDC) * 4;      // sHP  -> 137216 B
  constexpr size_t SMEM_B = (size_t)(TILE * LDP) * 2       // sP
                          + (size_t)(HD * LDP) * 2         // sV
                          + (size_t)(TILE * HD) * 2        // sVraw
                          + (size_t)(TILE * LDO) * 4       // sO
                          + (size_t)(4 * TILE) * 4         // sEs/sRM/sRS/sEd
                          + 256 * 4;                       // sRed -> 106496 B

  gat_hp_wmma<<<128, 256, SMEM_A, stream>>>(hg, Wg, ag, hp16, e_src, e_dst);
  gat_attn_wmma<<<32 * 8, 256, SMEM_B, stream>>>(hg, hp16, e_src, e_dst, out);
}
